// PointNet2Fast_85804856639963
// MI455X (gfx1250) — compile-verified, matrix operand order checked by reading, not run on hardware
//
#include <hip/hip_runtime.h>
#include <cstdint>
#include <cstddef>

typedef __attribute__((ext_vector_type(16))) _Float16 v16h;
typedef __attribute__((ext_vector_type(8)))  _Float16 v8h;
typedef __attribute__((ext_vector_type(8)))  float    v8f;

#define PN2_EPS 1e-5f

// ---------------------------------------------------------------------------
// Weight packing: f32 [Cout, Cin] -> f16 [Cout, ldp], zero padded in K.
// ---------------------------------------------------------------------------
__global__ void pn2_packw(const float* __restrict__ w, _Float16* __restrict__ o,
                          int Cout, int Cin, int ldp) {
  int t = blockIdx.x * blockDim.x + threadIdx.x;
  if (t >= Cout * ldp) return;
  int c = t % ldp, r = t / ldp;
  o[t] = (c < Cin) ? (_Float16)w[(size_t)r * Cin + c] : (_Float16)0.f;
}

// ---------------------------------------------------------------------------
// Brute-force kNN: per (b,n) scan all m, keep top-16 of pd = 2 x.y - |x|^2 - |y|^2
// Stable insertion (strict >) reproduces jax.lax.top_k tie order (lower idx first).
// ---------------------------------------------------------------------------
__global__ void pn2_knn(const float* __restrict__ x, int* __restrict__ out,
                        int Bn, int N, int C) {
  int t = blockIdx.x * blockDim.x + threadIdx.x;
  if (t >= Bn * N) return;
  int b = t / N, n = t % N;
  const float* xb = x + (size_t)b * N * C;
  float xl[132];
  float sqn = 0.f;
  for (int c = 0; c < C; ++c) { float v = xb[(size_t)n * C + c]; xl[c] = v; sqn += v * v; }
  float bd[16]; int bi[16];
  for (int j = 0; j < 16; ++j) { bd[j] = -3.4e38f; bi[j] = 0; }
  for (int m = 0; m < N; ++m) {
    const float* xm = xb + (size_t)m * C;
    float dot = 0.f, sqm = 0.f;
    for (int c = 0; c < C; ++c) { float v = xm[c]; dot += xl[c] * v; sqm += v * v; }
    float pd = 2.f * dot - sqn - sqm;
    if (pd > bd[15]) {
      int p = 15;
      while (p > 0 && pd > bd[p - 1]) { bd[p] = bd[p - 1]; bi[p] = bi[p - 1]; --p; }
      bd[p] = pd; bi[p] = m;
    }
  }
  for (int j = 0; j < 16; ++j) out[(size_t)t * 16 + j] = bi[j];
}

// ---------------------------------------------------------------------------
// Edge features: row = (b*N+n)*16+k -> [nb-ctr (C), ctr (C), 0-pad to ldp], f16
// ---------------------------------------------------------------------------
__global__ void pn2_edge(const float* __restrict__ x, const int* __restrict__ knn,
                         _Float16* __restrict__ h, int Bn, int N, int C, int ldp) {
  int row = blockIdx.x * blockDim.x + threadIdx.x;
  int M = Bn * N * 16;
  if (row >= M) return;
  int k = row & 15, bn = row >> 4;
  int b = bn / N, n = bn % N;
  int m = knn[(size_t)bn * 16 + k];
  const float* ctr = x + ((size_t)b * N + n) * C;
  const float* nb  = x + ((size_t)b * N + m) * C;
  _Float16* o = h + (size_t)row * ldp;
  for (int c = 0; c < C; ++c) o[c]     = (_Float16)(nb[c] - ctr[c]);
  for (int c = 0; c < C; ++c) o[C + c] = (_Float16)ctr[c];
  for (int c = 2 * C; c < ldp; ++c) o[c] = (_Float16)0.f;
}

// ---------------------------------------------------------------------------
// WMMA f16 GEMM, 16x64 macro-tile per wave (A fragment reused across 4 N-tiles):
//   Y[M,Cout] = H[M,K] x W[Cout,K]^T   (K padded to mult of 32, Cout mult of 64)
// A and B frags share the 16-bit 16x32 per-lane layout (ISA 7.12.2):
// lane l holds row (l&15), K chunks [8*(l>=16) .. +8) and [+16 ..).
// ---------------------------------------------------------------------------
__device__ __forceinline__ v16h pn2_load_frag(const _Float16* __restrict__ base,
                                              int ld, int lane, int kt) {
  int r  = lane & 15;
  int kb = (lane >> 4) << 3;
  const _Float16* p = base + (size_t)r * ld + (size_t)kt * 32 + kb;
  v8h lo = *(const v8h*)(p);
  v8h hi = *(const v8h*)(p + 16);
  v16h f;
#pragma unroll
  for (int i = 0; i < 8; ++i) { f[i] = lo[i]; f[i + 8] = hi[i]; }
  return f;
}

__device__ __forceinline__ void pn2_store_tile(_Float16* __restrict__ Y, int Cout,
                                               long long tm, int ncol0, int lane,
                                               const v8f& acc) {
  int col = lane & 15;
  int rb  = (lane >> 4) << 3;   // C/D layout: VGPR r -> row r + 8*(lane>=16)
  _Float16* yb = Y + ((size_t)tm * 16 + rb) * Cout + (size_t)ncol0 + col;
#pragma unroll
  for (int r = 0; r < 8; ++r) yb[(size_t)r * Cout] = (_Float16)acc[r];
}

__global__ void pn2_gemm_wmma(const _Float16* __restrict__ H,
                              const _Float16* __restrict__ W,
                              _Float16* __restrict__ Y,
                              int M, int K, int Cout) {
  int wave = (int)(threadIdx.x >> 5);
  int lane = (int)(threadIdx.x & 31);
  int tilesN = Cout >> 6;            // 64-wide macro tiles
  long long gid   = (long long)blockIdx.x * (blockDim.x >> 5) + wave;
  long long total = (long long)(M >> 4) * tilesN;
  if (gid >= total) return;          // wave-uniform: EXEC stays all-ones for WMMA
  int tn = (int)(gid % tilesN);
  long long tm = gid / tilesN;
  const _Float16* hb = H + (size_t)tm * 16 * K;
  const _Float16* wb = W + (size_t)tn * 64 * K;
  v8f acc0 = {}, acc1 = {}, acc2 = {}, acc3 = {};
  int nk = K >> 5;
  for (int kt = 0; kt < nk; ++kt) {
    v16h a  = pn2_load_frag(hb, K, lane, kt);
    v16h b0 = pn2_load_frag(wb + (size_t)0  * K, K, lane, kt);
    v16h b1 = pn2_load_frag(wb + (size_t)16 * K, K, lane, kt);
    v16h b2 = pn2_load_frag(wb + (size_t)32 * K, K, lane, kt);
    v16h b3 = pn2_load_frag(wb + (size_t)48 * K, K, lane, kt);
    acc0 = __builtin_amdgcn_wmma_f32_16x16x32_f16(false, a, false, b0, (short)0, acc0, false, false);
    acc1 = __builtin_amdgcn_wmma_f32_16x16x32_f16(false, a, false, b1, (short)0, acc1, false, false);
    acc2 = __builtin_amdgcn_wmma_f32_16x16x32_f16(false, a, false, b2, (short)0, acc2, false, false);
    acc3 = __builtin_amdgcn_wmma_f32_16x16x32_f16(false, a, false, b3, (short)0, acc3, false, false);
  }
  pn2_store_tile(Y, Cout, tm, tn * 64 + 0,  lane, acc0);
  pn2_store_tile(Y, Cout, tm, tn * 64 + 16, lane, acc1);
  pn2_store_tile(Y, Cout, tm, tn * 64 + 32, lane, acc2);
  pn2_store_tile(Y, Cout, tm, tn * 64 + 48, lane, acc3);
}

// ---------------------------------------------------------------------------
// Per-channel sum / sumsq over M rows (block-reduce + atomics).
// ---------------------------------------------------------------------------
__global__ void pn2_colstats(const _Float16* __restrict__ y, int M, int C,
                             float* __restrict__ sum, float* __restrict__ sumsq) {
  __shared__ float s1[256], s2[256];
  int c = blockIdx.x;
  float a = 0.f, a2 = 0.f;
  for (int r = blockIdx.y * blockDim.x + threadIdx.x; r < M;
       r += blockDim.x * gridDim.y) {
    float v = (float)y[(size_t)r * C + c];
    a += v; a2 += v * v;
  }
  s1[threadIdx.x] = a; s2[threadIdx.x] = a2;
  __syncthreads();
  for (int s = 128; s > 0; s >>= 1) {
    if ((int)threadIdx.x < s) { s1[threadIdx.x] += s1[threadIdx.x + s];
                                s2[threadIdx.x] += s2[threadIdx.x + s]; }
    __syncthreads();
  }
  if (threadIdx.x == 0) { atomicAdd(&sum[c], s1[0]); atomicAdd(&sumsq[c], s2[0]); }
}

__global__ void pn2_bnfinal(const float* __restrict__ sum, const float* __restrict__ sumsq,
                            const float* __restrict__ g, const float* __restrict__ bb,
                            float invM, float* __restrict__ scale,
                            float* __restrict__ shift, int C) {
  int c = blockIdx.x * blockDim.x + threadIdx.x;
  if (c >= C) return;
  float mu  = sum[c] * invM;
  float var = fmaxf(sumsq[c] * invM - mu * mu, 0.f);
  float s = g[c] * rsqrtf(var + PN2_EPS);
  scale[c] = s;
  shift[c] = bb[c] - mu * s;
}

__global__ void pn2_bnrelu(_Float16* __restrict__ y, long long total, int C,
                           const float* __restrict__ scale,
                           const float* __restrict__ shift) {
  size_t e = (size_t)blockIdx.x * blockDim.x + threadIdx.x;
  if (e >= (size_t)total) return;
  int c = (int)(e % (size_t)C);
  float v = (float)y[e] * scale[c] + shift[c];
  y[e] = (_Float16)fmaxf(v, 0.f);
}

// ---------------------------------------------------------------------------
// Fused BN + ReLU + max over k=16 + subsample: out[b,j,off+c] (f32, stride outld)
// ---------------------------------------------------------------------------
__global__ void pn2_poolsel(const _Float16* __restrict__ y,
                            const float* __restrict__ scale,
                            const float* __restrict__ shift,
                            const int* __restrict__ idx, float* __restrict__ out,
                            int Bn, int Nsrc, int Nsel, int C, int outld, int outoff) {
  int t = blockIdx.x * blockDim.x + threadIdx.x;
  if (t >= Bn * Nsel * C) return;
  int c = t % C, j = (t / C) % Nsel, b = t / (C * Nsel);
  int n = idx[j];
  size_t base = ((size_t)b * Nsrc + n) * 16 * (size_t)C + c;
  float sm = -3.4e38f;
  for (int k = 0; k < 16; ++k)
    sm = fmaxf(sm, (float)y[base + (size_t)k * C] * scale[c] + shift[c]);
  out[((size_t)b * Nsel + j) * outld + outoff + c] = fmaxf(sm, 0.f);
}

__global__ void pn2_gather3(const float* __restrict__ src, int srcld,
                            const int* __restrict__ idx, float* __restrict__ dst,
                            int dstld, int Bn, int Nsrc, int Nsel) {
  int t = blockIdx.x * blockDim.x + threadIdx.x;
  if (t >= Bn * Nsel * 3) return;
  int d = t % 3, j = (t / 3) % Nsel, b = t / (3 * Nsel);
  dst[((size_t)b * Nsel + j) * dstld + d] =
      src[((size_t)b * Nsrc + idx[j]) * srcld + d];
}

__global__ void pn2_f32tof16(const float* __restrict__ src, _Float16* __restrict__ o,
                             int M, int C, int ldp) {
  int t = blockIdx.x * blockDim.x + threadIdx.x;
  if (t >= M * ldp) return;
  int c = t % ldp, r = t / ldp;
  o[t] = (c < C) ? (_Float16)src[(size_t)r * C + c] : (_Float16)0.f;
}

// BN + ReLU + max over the N points of each batch: pooled[b,c]
__global__ void pn2_rowpool(const _Float16* __restrict__ y,
                            const float* __restrict__ scale,
                            const float* __restrict__ shift,
                            float* __restrict__ out, int Bn, int N, int C) {
  int t = blockIdx.x * blockDim.x + threadIdx.x;
  if (t >= Bn * C) return;
  int c = t % C, b = t / C;
  float sm = -3.4e38f;
  for (int n = 0; n < N; ++n)
    sm = fmaxf(sm, (float)y[((size_t)b * N + n) * C + c] * scale[c] + shift[c]);
  out[(size_t)b * C + c] = fmaxf(sm, 0.f);
}

// ---------------------------------------------------------------------------
// Tiny FC head (8 rows): plain f32 dot products + BN over batch axis.
// ---------------------------------------------------------------------------
__global__ void pn2_fc(const float* __restrict__ h, const float* __restrict__ w,
                       float* __restrict__ y, int Bn, int Cin, int Cout) {
  int t = blockIdx.x * blockDim.x + threadIdx.x;
  if (t >= Bn * Cout) return;
  int o = t % Cout, b = t / Cout;
  float acc = 0.f;
  for (int c = 0; c < Cin; ++c) acc += h[(size_t)b * Cin + c] * w[(size_t)o * Cin + c];
  y[t] = acc;
}

__global__ void pn2_bnfc(const float* __restrict__ y, const float* __restrict__ g,
                         const float* __restrict__ bb, float* __restrict__ h,
                         int Bn, int C) {
  int c = blockIdx.x * blockDim.x + threadIdx.x;
  if (c >= C) return;
  float mu = 0.f;
  for (int b = 0; b < Bn; ++b) mu += y[(size_t)b * C + c];
  mu /= (float)Bn;
  float var = 0.f;
  for (int b = 0; b < Bn; ++b) { float d = y[(size_t)b * C + c] - mu; var += d * d; }
  var /= (float)Bn;
  float s = g[c] * rsqrtf(var + PN2_EPS);
  float sh = bb[c] - mu * s;
  for (int b = 0; b < Bn; ++b)
    h[(size_t)b * C + c] = fmaxf(y[(size_t)b * C + c] * s + sh, 0.f);
}

__global__ void pn2_fc3(const float* __restrict__ h, const float* __restrict__ w,
                        const float* __restrict__ bias, float* __restrict__ out,
                        int Bn, int Cin, int Cout) {
  int t = blockIdx.x * blockDim.x + threadIdx.x;
  if (t >= Bn * Cout) return;
  int o = t % Cout, b = t / Cout;
  float acc = bias[o];
  for (int c = 0; c < Cin; ++c) acc += h[(size_t)b * Cin + c] * w[(size_t)o * Cin + c];
  out[t] = acc;
}

// ---------------------------------------------------------------------------
// Orchestration
// ---------------------------------------------------------------------------
extern "C" void kernel_launch(void* const* d_in, const int* in_sizes, int n_in,
                              void* d_out, int out_size, void* d_ws, size_t ws_size,
                              hipStream_t stream) {
  (void)in_sizes; (void)n_in; (void)out_size; (void)ws_size;
  const int Bn = 8, N1 = 4096, N2 = 512, N3 = 128;

  // Input order assumption: x, idx1, idx2, then params in insertion order
  // (sa1_0..fc2 each as w,g,b) then fc3_w, fc3_b.
  const float* X = (const float*)d_in[0];
  const int* I1  = (const int*)d_in[1];
  const int* I2  = (const int*)d_in[2];
  auto Wp = [&](int li) { return (const float*)d_in[3 + 3 * li + 0]; };
  auto Gp = [&](int li) { return (const float*)d_in[3 + 3 * li + 1]; };
  auto Bp = [&](int li) { return (const float*)d_in[3 + 3 * li + 2]; };
  const float* fc3w = (const float*)d_in[36];
  const float* fc3b = (const float*)d_in[37];

  struct LayerD { int Cin, Cp, Cout; };
  const LayerD LD[9] = {{6, 32, 64},    {64, 64, 64},   {64, 64, 128},
                        {262, 288, 128},{128, 128, 128},{128, 128, 256},
                        {259, 288, 256},{256, 256, 512},{512, 512, 1024}};

  char* ws = (char*)d_ws;
  size_t off = 0;
  auto alloc = [&](size_t bytes) -> char* {
    off = (off + 255) & ~(size_t)255;
    char* p = ws + off;
    off += bytes;
    return p;
  };

  const size_t M1 = (size_t)Bn * N1 * 16;  // 524288 edge rows
  const size_t M2 = (size_t)Bn * N2 * 16;  // 65536
  const size_t M3 = (size_t)Bn * N3;       // 1024

  _Float16* bufA = (_Float16*)alloc(M1 * 64 * 2);    // 64 MB (max ld among A users)
  _Float16* bufB = (_Float16*)alloc(M1 * 128 * 2);   // 128 MB
  _Float16* wp[9];
  for (int i = 0; i < 9; ++i) wp[i] = (_Float16*)alloc((size_t)LD[i].Cout * LD[i].Cp * 2);
  int*   knn1   = (int*)alloc((size_t)Bn * N1 * 16 * 4);
  int*   knn2   = (int*)alloc((size_t)Bn * N2 * 16 * 4);
  float* x1     = (float*)alloc((size_t)Bn * N2 * 131 * 4);
  float* x3     = (float*)alloc((size_t)Bn * N3 * 259 * 4);
  float* pooled = (float*)alloc((size_t)Bn * 1024 * 4);
  float* yfc1   = (float*)alloc((size_t)Bn * 512 * 4);
  float* hfc1   = (float*)alloc((size_t)Bn * 512 * 4);
  float* yfc2   = (float*)alloc((size_t)Bn * 256 * 4);
  float* hfc2   = (float*)alloc((size_t)Bn * 256 * 4);
  float* ssum   = (float*)alloc(1024 * 4);
  float* ssq    = (float*)alloc(1024 * 4);
  float* scl    = (float*)alloc(1024 * 4);
  float* shf    = (float*)alloc(1024 * 4);

  auto cdiv = [](long long a, long long b) { return (int)((a + b - 1) / b); };

  for (int i = 0; i < 9; ++i) {
    int tot = LD[i].Cout * LD[i].Cp;
    pn2_packw<<<cdiv(tot, 256), 256, 0, stream>>>(Wp(i), wp[i], LD[i].Cout,
                                                  LD[i].Cin, LD[i].Cp);
  }

  auto conv = [&](const _Float16* Hin, _Float16* Yout, size_t M, int li, bool act) {
    const int K = LD[li].Cp, Cout = LD[li].Cout;
    long long tiles = (long long)(M >> 4) * (Cout >> 6);  // 16x64 macro-tiles
    pn2_gemm_wmma<<<cdiv(tiles, 8), 256, 0, stream>>>(Hin, wp[li], Yout,
                                                      (int)M, K, Cout);
    hipMemsetAsync(ssum, 0, (size_t)Cout * 4, stream);
    hipMemsetAsync(ssq,  0, (size_t)Cout * 4, stream);
    int parts = cdiv((long long)M, 256); if (parts > 64) parts = 64;
    dim3 g((unsigned)Cout, (unsigned)parts);
    pn2_colstats<<<g, 256, 0, stream>>>(Yout, (int)M, Cout, ssum, ssq);
    pn2_bnfinal<<<cdiv(Cout, 256), 256, 0, stream>>>(ssum, ssq, Gp(li), Bp(li),
                                                     1.f / (float)M, scl, shf, Cout);
    if (act) {
      long long tot = (long long)M * Cout;
      pn2_bnrelu<<<cdiv(tot, 256), 256, 0, stream>>>(Yout, tot, Cout, scl, shf);
    }
  };

  // ---- stage 1: knn on xyz, edge feats (C=3 -> 6, pad 32), convs 64,64,128 ----
  pn2_knn<<<cdiv(Bn * N1, 128), 128, 0, stream>>>(X, knn1, Bn, N1, 3);
  pn2_edge<<<cdiv((long long)M1, 256), 256, 0, stream>>>(X, knn1, bufA, Bn, N1, 3, 32);
  conv(bufA, bufB, M1, 0, true);
  conv(bufB, bufA, M1, 1, true);
  conv(bufA, bufB, M1, 2, false);
  pn2_gather3<<<cdiv(Bn * N2 * 3, 256), 256, 0, stream>>>(X, 3, I1, x1, 131, Bn, N1, N2);
  pn2_poolsel<<<cdiv(Bn * N2 * 128, 256), 256, 0, stream>>>(bufB, scl, shf, I1, x1,
                                                            Bn, N1, N2, 128, 131, 3);

  // ---- stage 2: knn on 131-d, edge feats (262, pad 288), convs 128,128,256 ----
  pn2_knn<<<cdiv(Bn * N2, 128), 128, 0, stream>>>(x1, knn2, Bn, N2, 131);
  pn2_edge<<<cdiv((long long)M2, 256), 256, 0, stream>>>(x1, knn2, bufA, Bn, N2, 131, 288);
  conv(bufA, bufB, M2, 3, true);
  conv(bufB, bufA, M2, 4, true);
  conv(bufA, bufB, M2, 5, false);
  pn2_gather3<<<cdiv(Bn * N3 * 3, 256), 256, 0, stream>>>(x1, 131, I2, x3, 259, Bn, N2, N3);
  pn2_poolsel<<<cdiv(Bn * N3 * 256, 256), 256, 0, stream>>>(bufB, scl, shf, I2, x3,
                                                            Bn, N2, N3, 256, 259, 3);

  // ---- stage 3: pointwise convs 256,512,1024 on (B*128, 259 pad 288) ----
  pn2_f32tof16<<<cdiv((long long)M3 * 288, 256), 256, 0, stream>>>(x3, bufA,
                                                                   (int)M3, 259, 288);
  conv(bufA, bufB, M3, 6, true);
  conv(bufB, bufA, M3, 7, true);
  conv(bufA, bufB, M3, 8, false);
  pn2_rowpool<<<cdiv(Bn * 1024, 256), 256, 0, stream>>>(bufB, scl, shf, pooled,
                                                        Bn, N3, 1024);

  // ---- fc head ----
  pn2_fc<<<cdiv(Bn * 512, 256), 256, 0, stream>>>(pooled, Wp(9), yfc1, Bn, 1024, 512);
  pn2_bnfc<<<cdiv(512, 256), 256, 0, stream>>>(yfc1, Gp(9), Bp(9), hfc1, Bn, 512);
  pn2_fc<<<cdiv(Bn * 256, 256), 256, 0, stream>>>(hfc1, Wp(10), yfc2, Bn, 512, 256);
  pn2_bnfc<<<cdiv(256, 256), 256, 0, stream>>>(yfc2, Gp(10), Bp(10), hfc2, Bn, 256);
  pn2_fc3<<<1, 64, 0, stream>>>(hfc2, fc3w, fc3b, (float*)d_out, Bn, 256, 2);
}
